// Quantiser_26061861552625
// MI455X (gfx1250) — matrix operand order repaired and checked by reading, not compiled, not run on hardware
//
#include <hip/hip_runtime.h>

// ---------------------------------------------------------------------------
// VQ quantiser for MI455X (gfx1250, wave32, WMMA).
// Split-bf16 GEMM (hi+lo, fp32 accumulate) via v_wmma_f32_16x16x32_bf16,
// fused w^2 - 2*x.wT argmin, double-buffered LDS W-tiles filled with
// GLOBAL_LOAD_ASYNC_TO_LDS_B128 (ASYNCcnt), then gather + loss.
// ---------------------------------------------------------------------------

typedef __attribute__((ext_vector_type(16))) __bf16 v16bf;
typedef __attribute__((ext_vector_type(8)))  float  v8f;

#define NPTS 32768
#define DDIM 256
#define VCODES 8192

// Builtin signature (from hipcc diagnostic): (v4i AS1*, v4i AS3*, int, int)
typedef int v4i __attribute__((vector_size(16)));
typedef __attribute__((address_space(1))) v4i GAS_v4i;   // global
typedef __attribute__((address_space(3))) v4i LAS_v4i;   // LDS

union FragU { uint4 q[2]; v16bf v; };

static __device__ __forceinline__ v8f wmma_bf16(v16bf a, v16bf b, v8f c) {
    // (neg_a, A, neg_b, B, c_mod, C, reuse_a, reuse_b)
    return __builtin_amdgcn_wmma_f32_16x16x32_bf16(false, a, false, b,
                                                   (short)0, c, false, false);
}

static __device__ __forceinline__ void async_ld_b128(const void* g, void* l) {
#if __has_builtin(__builtin_amdgcn_global_load_async_to_lds_b128)
    __builtin_amdgcn_global_load_async_to_lds_b128((GAS_v4i*)g, (LAS_v4i*)l,
                                                   0, 0);
#else
    unsigned int lofs = (unsigned int)(unsigned long long)(LAS_v4i*)l;
    asm volatile("global_load_async_to_lds_b128 %0, %1, off"
                 :: "v"(lofs), "v"(g) : "memory");
#endif
}

static __device__ __forceinline__ void wait_async0() {
#if __has_builtin(__builtin_amdgcn_s_wait_asynccnt)
    __builtin_amdgcn_s_wait_asynccnt(0);
#else
    asm volatile("s_wait_asynccnt 0x0" ::: "memory");
#endif
}

// ---------------------------------------------------------------------------
// Kernel 1: fp32 -> (bf16 hi, bf16 lo) split conversion, 8 elems / thread.
// ---------------------------------------------------------------------------
__global__ __launch_bounds__(256) void k_split_bf16(
    const float* __restrict__ src, unsigned short* __restrict__ hi_us,
    unsigned short* __restrict__ lo_us, int n8) {
    int t = blockIdx.x * 256 + threadIdx.x;
    if (t >= n8) return;
    const float4* s = (const float4*)src;
    float4 a = s[t * 2 + 0];
    float4 b = s[t * 2 + 1];
    float f[8] = {a.x, a.y, a.z, a.w, b.x, b.y, b.z, b.w};
    union { __bf16 h[8]; uint4 q; } uh, ul;
#pragma unroll
    for (int i = 0; i < 8; ++i) {
        __bf16 h = (__bf16)f[i];
        uh.h[i]  = h;
        ul.h[i]  = (__bf16)(f[i] - (float)h);
    }
    ((uint4*)hi_us)[t] = uh.q;
    ((uint4*)lo_us)[t] = ul.q;
}

// ---------------------------------------------------------------------------
// Kernel 2: w2[v] = sum_k W[v,k]^2 ; one wave per codebook row.
// Also zeroes the loss accumulator (runs before the gather kernel).
// ---------------------------------------------------------------------------
__global__ __launch_bounds__(256) void k_w2(const float* __restrict__ W,
                                            float* __restrict__ w2,
                                            float* __restrict__ lossAcc) {
    if (blockIdx.x == 0 && threadIdx.x == 0) *lossAcc = 0.0f;
    const int wv = threadIdx.x >> 5, lane = threadIdx.x & 31;
    const int row = blockIdx.x * 8 + wv;
    const float4* wr = (const float4*)(W + (size_t)row * DDIM);
    float4 a = wr[lane * 2 + 0];
    float4 b = wr[lane * 2 + 1];
    float s = a.x * a.x + a.y * a.y + a.z * a.z + a.w * a.w +
              b.x * b.x + b.y * b.y + b.z * b.z + b.w * b.w;
#pragma unroll
    for (int m = 16; m >= 1; m >>= 1) s += __shfl_xor(s, m, 32);
    if (lane == 0) w2[row] = s;
}

// ---------------------------------------------------------------------------
// Async prefetch of one 32-code W tile (hi+lo) into an LDS buffer.
// Buffer layout: [mat(2)][row(32)][k(256)] bf16 = 2048 uint4 = 32KB.
// 8 b128 async loads per thread (256 threads).
// ---------------------------------------------------------------------------
static __device__ __forceinline__ void prefetch_tile(
    const __bf16* __restrict__ wbase, char* ldsBase, int vbase, int t) {
#pragma unroll
    for (int r = 0; r < 8; ++r) {
        const int i   = t + r * 256;
        const int mat = i >> 10;
        const int row = (i >> 5) & 31;
        const int kq  = i & 31;
        const __bf16* g = wbase + (size_t)mat * ((size_t)VCODES * DDIM)
                                + (size_t)(vbase + row) * DDIM + (size_t)kq * 8;
        async_ld_b128((const void*)g, (void*)(ldsBase + (size_t)i * 16));
    }
}

// ---------------------------------------------------------------------------
// Kernel 3: fused distance + argmin.
//   score[n,v] = w2[v] - 2 * (x[n,:] . W[v,:])     (x^2 dropped: row-constant)
// Block = 256 thr = 8 waves; block owns 128 rows of X; wave owns 16 rows.
// A fragments (hi+lo, full K=256) live in VGPRs for the whole V sweep.
// W tiles (32 codes, hi+lo) double-buffered in LDS via async loads.
// ---------------------------------------------------------------------------
__global__ __launch_bounds__(256) void k_argmin(
    const unsigned short* __restrict__ xhi_us, const unsigned short* __restrict__ xlo_us,
    const unsigned short* __restrict__ whi_us, const unsigned short* __restrict__ wlo_us,
    const float* __restrict__ w2g, int* __restrict__ idxOut) {
    __shared__ uint4 ldsq[4096];   // 2 x 32KB double buffer

    const __bf16* xhi = (const __bf16*)xhi_us;
    const __bf16* xlo = (const __bf16*)xlo_us;
    const __bf16* whi = (const __bf16*)whi_us;   // wlo is contiguous after whi
    (void)wlo_us;

    const int t = threadIdx.x;
    const int wv = t >> 5, lane = t & 31;
    const int rowBase = blockIdx.x * 128 + wv * 16;

    // ---- A fragments: row = rowBase + lane%16; ISA 16-bit A 16x32 layout:
    // lane<16 -> K in {0..7,16..23}; lane>=16 -> K in {8..15,24..31}.
    const int arow   = rowBase + (lane & 15);
    const int kA4    = (lane < 16) ? 0 : 1;  // kbase/8 in uint4 units
    const uint4* xhr = (const uint4*)(xhi + (size_t)arow * DDIM);
    const uint4* xlr = (const uint4*)(xlo + (size_t)arow * DDIM);
    v16bf Ahi[8], Alo[8];
#pragma unroll
    for (int ks = 0; ks < 8; ++ks) {
        const int q0 = ks * 4 + kA4;
        FragU u; u.q[0] = xhr[q0]; u.q[1] = xhr[q0 + 2]; Ahi[ks] = u.v;
        FragU v; v.q[0] = xlr[q0]; v.q[1] = xlr[q0 + 2]; Alo[ks] = v.v;
    }

    float best[8];
    int   bidx[8];
#pragma unroll
    for (int j = 0; j < 8; ++j) { best[j] = 3.4e38f; bidx[j] = 0; }

    // ---- B fragment addressing (32x16 B layout): lane<16 -> K {0..7,16..23},
    // lane>=16 -> K {8..15,24..31}. Row stride = 32 uint4; mat offset = 1024.
    const int kB4 = (lane < 16) ? 0 : 1;
    const int bH0 = ((lane & 15) + 0)  * 32;          // subtile 0, hi
    const int bH1 = ((lane & 15) + 16) * 32;          // subtile 1, hi
    const int bL0 = 1024 + ((lane & 15) + 0)  * 32;   // subtile 0, lo
    const int bL1 = 1024 + ((lane & 15) + 16) * 32;   // subtile 1, lo

    prefetch_tile(whi, (char*)ldsq, 0, t);

    const int NVT = VCODES / 32;   // 256 iterations
#pragma unroll 1
    for (int vt = 0; vt < NVT; ++vt) {
        wait_async0();        // our buffer's async fills are complete
        __syncthreads();      // everyone's fills done + prev reads of other buf done
        if (vt + 1 < NVT)
            prefetch_tile(whi, (char*)ldsq + (size_t)((vt + 1) & 1) * 32768,
                          (vt + 1) * 32, t);

        const int bufq  = (vt & 1) * 2048;
        const int vbase = vt * 32;
        const float w2v0 = w2g[vbase + (lane & 15)];
        const float w2v1 = w2g[vbase + 16 + (lane & 15)];

        v8f acc0 = {0.f, 0.f, 0.f, 0.f, 0.f, 0.f, 0.f, 0.f};
        v8f acc1 = {0.f, 0.f, 0.f, 0.f, 0.f, 0.f, 0.f, 0.f};
#pragma unroll
        for (int ks = 0; ks < 8; ++ks) {
            const int q0 = ks * 4 + kB4;
            FragU bh0, bh1, bl0, bl1;
            bh0.q[0] = ldsq[bufq + bH0 + q0]; bh0.q[1] = ldsq[bufq + bH0 + q0 + 2];
            bh1.q[0] = ldsq[bufq + bH1 + q0]; bh1.q[1] = ldsq[bufq + bH1 + q0 + 2];
            bl0.q[0] = ldsq[bufq + bL0 + q0]; bl0.q[1] = ldsq[bufq + bL0 + q0 + 2];
            bl1.q[0] = ldsq[bufq + bL1 + q0]; bl1.q[1] = ldsq[bufq + bL1 + q0 + 2];
            acc0 = wmma_bf16(Ahi[ks], bh0.v, acc0);   // hi*hi   (chain 0)
            acc1 = wmma_bf16(Ahi[ks], bh1.v, acc1);   // hi*hi   (chain 1)
            acc0 = wmma_bf16(Ahi[ks], bl0.v, acc0);   // hi*lo
            acc1 = wmma_bf16(Ahi[ks], bl1.v, acc1);
            acc0 = wmma_bf16(Alo[ks], bh0.v, acc0);   // lo*hi
            acc1 = wmma_bf16(Alo[ks], bh1.v, acc1);
        }

        const int cand0 = vbase + (lane & 15);
        const int cand1 = vbase + 16 + (lane & 15);
#pragma unroll
        for (int j = 0; j < 8; ++j) {
            const float sc0 = fmaf(-2.0f, acc0[j], w2v0);
            if (sc0 < best[j]) { best[j] = sc0; bidx[j] = cand0; }
            const float sc1 = fmaf(-2.0f, acc1[j], w2v1);
            if (sc1 < best[j]) { best[j] = sc1; bidx[j] = cand1; }
        }
    }

    // cross-lane min within each 16-lane half (C/D layout: N = lane%16,
    // M = vgpr + 8*(lane>=16)); tie-break to lowest index (argmin semantics)
#pragma unroll
    for (int j = 0; j < 8; ++j) {
#pragma unroll
        for (int m = 1; m < 16; m <<= 1) {
            const float os = __shfl_xor(best[j], m, 32);
            const int   oi = __shfl_xor(bidx[j], m, 32);
            if (os < best[j] || (os == best[j] && oi < bidx[j])) {
                best[j] = os; bidx[j] = oi;
            }
        }
    }
    if (lane == 0) {
#pragma unroll
        for (int j = 0; j < 8; ++j) idxOut[rowBase + j] = bidx[j];
    } else if (lane == 16) {
#pragma unroll
        for (int j = 0; j < 8; ++j) idxOut[rowBase + 8 + j] = bidx[j];
    }
}

// ---------------------------------------------------------------------------
// Kernel 4: gather quantised rows, write float(idx), accumulate sum (q-x)^2.
// One wave per row.
// ---------------------------------------------------------------------------
__global__ __launch_bounds__(256) void k_gather(
    const float* __restrict__ X, const float* __restrict__ W,
    const int* __restrict__ idx, float* __restrict__ outQ,
    float* __restrict__ outIdxF, float* __restrict__ lossAcc) {
    const int wv = threadIdx.x >> 5, lane = threadIdx.x & 31;
    const int r  = blockIdx.x * 8 + wv;
    const int id = idx[r];
    const float4* qr = (const float4*)(W + (size_t)id * DDIM);
    const float4* xr = (const float4*)(X + (size_t)r * DDIM);
    float4 q0 = qr[lane * 2 + 0], q1 = qr[lane * 2 + 1];
    float4 x0 = xr[lane * 2 + 0], x1 = xr[lane * 2 + 1];
    ((float4*)(outQ + (size_t)r * DDIM))[lane * 2 + 0] = q0;
    ((float4*)(outQ + (size_t)r * DDIM))[lane * 2 + 1] = q1;
    float d, s = 0.f;
    d = q0.x - x0.x; s += d * d;  d = q0.y - x0.y; s += d * d;
    d = q0.z - x0.z; s += d * d;  d = q0.w - x0.w; s += d * d;
    d = q1.x - x1.x; s += d * d;  d = q1.y - x1.y; s += d * d;
    d = q1.z - x1.z; s += d * d;  d = q1.w - x1.w; s += d * d;
#pragma unroll
    for (int m = 16; m >= 1; m >>= 1) s += __shfl_xor(s, m, 32);
    if (lane == 0) {
        atomicAdd(lossAcc, s);
        outIdxF[r] = (float)id;
    }
}

// ---------------------------------------------------------------------------
// Kernel 5: loss = (1 + 0.25) * sum / (N*D)   (forward values of both loss
// terms are identical; stop_gradient only affects backward).
// ---------------------------------------------------------------------------
__global__ void k_finalize(const float* __restrict__ lossAcc,
                           float* __restrict__ outLoss) {
    *outLoss = 1.25f * (*lossAcc) / (float)((size_t)NPTS * DDIM);
}

// ---------------------------------------------------------------------------
extern "C" void kernel_launch(void* const* d_in, const int* in_sizes, int n_in,
                              void* d_out, int out_size, void* d_ws, size_t ws_size,
                              hipStream_t stream) {
    (void)in_sizes; (void)n_in; (void)out_size; (void)ws_size;
    const float* x = (const float*)d_in[0];        // [N, D] fp32
    const float* W = (const float*)d_in[1];        // [V, D] fp32
    float* out = (float*)d_out;                    // [N*D] q | [1] loss | [N] idx

    const size_t ND = (size_t)NPTS * DDIM;         // 8,388,608
    const size_t VD = (size_t)VCODES * DDIM;       // 2,097,152

    // workspace layout (~40.2 MB); wlo MUST be contiguous after whi
    unsigned short* xhi = (unsigned short*)d_ws;
    unsigned short* xlo = xhi + ND;
    unsigned short* whi = xlo + ND;
    unsigned short* wlo = whi + VD;
    float* w2   = (float*)(wlo + VD);
    int*   idx  = (int*)(w2 + VCODES);
    float* lacc = (float*)(idx + NPTS);

    k_split_bf16<<<(int)(ND / 2048), 256, 0, stream>>>(x, xhi, xlo, (int)(ND / 8));
    k_split_bf16<<<(int)(VD / 2048), 256, 0, stream>>>(W, whi, wlo, (int)(VD / 8));
    k_w2<<<VCODES / 8, 256, 0, stream>>>(W, w2, lacc);
    k_argmin<<<NPTS / 128, 256, 0, stream>>>(xhi, xlo, whi, wlo, w2, idx);
    k_gather<<<NPTS / 8, 256, 0, stream>>>(x, W, idx, out, out + ND + 1, lacc);
    k_finalize<<<1, 1, 0, stream>>>(lacc, out + ND);
}
